// CandidateRegionGenerator_21534966022717
// MI455X (gfx1250) — matrix operand order repaired and checked by reading, not compiled, not run on hardware
//
#include <hip/hip_runtime.h>

typedef __attribute__((ext_vector_type(16))) _Float16 v16h;
typedef __attribute__((ext_vector_type(8)))  float    v8f;
typedef _Float16 h16;

#define BSZ   8
#define HH    128
#define WW    128
#define HWT   (HH*WW)          // 16384 = 2^14
#define KC    5
#define KITERS 10
#define EPSV  1e-5f

// ---------------------------------------------------------------- f32 -> f16
__global__ void k_cvt_f16(const float* __restrict__ src, h16* __restrict__ dst, int n) {
  int i = blockIdx.x * blockDim.x + threadIdx.x;
  if (i < n) dst[i] = (h16)src[i];
}

// ------------------------------------------------- pack weights to A-fragments
// ISA 16-bit A 16x32 layout: lane m = L%16, half group hh = L/16,
// half i<8 -> K = hh*8+i ; i>=8 -> K = 16 + hh*8 + (i-8).
// chunk = koff*(CIN/32) + ci_blk  (koff = kh*3+kw)
template<int CIN, int COUT>
__global__ void k_pack_w(const float* __restrict__ w, h16* __restrict__ wp) {
  const int NCB = CIN / 32;
  const int NCHUNK = 9 * NCB;
  const int total = (COUT / 16) * NCHUNK * 32;
  int gid = blockIdx.x * blockDim.x + threadIdx.x;
  if (gid >= total) return;
  int lane = gid & 31;
  int frag = gid >> 5;
  int chunk = frag % NCHUNK;
  int cot = frag / NCHUNK;
  int koff = chunk / NCB;
  int cib = chunk - koff * NCB;
  int m = lane & 15, hh = lane >> 4;
  h16* dst = wp + (size_t)gid * 16;
#pragma unroll
  for (int i = 0; i < 16; i++) {
    int kl = (i < 8) ? (hh * 8 + i) : (16 + hh * 8 + (i - 8));
    int ci = cib * 32 + kl;
    dst[i] = (h16)w[((size_t)(cot * 16 + m) * CIN + ci) * 9 + koff];
  }
}

// -------------------------------------------------------- implicit-GEMM conv
// Block = (batch bb, row y0, 64-pixel segment). Waves = COUT/16; each wave owns
// four 16x16 D tiles (4 accumulators) so every A-fragment load feeds 4 WMMAs.
// LDS patch layout [row(3)][px(66)][ci] in f16 -> B fragments read in two 16B runs
// (lane n = pixel, K = hh*8+i for i<8, 16+hh*8+(i-8) for i>=8 within the 32-channel chunk).
template<int CIN, int COUT>
__global__ void __launch_bounds__(COUT * 2)
k_conv(const h16* __restrict__ xh, const h16* __restrict__ wp,
       const float* __restrict__ bias, float* __restrict__ outp) {
  const int NCB = CIN / 32;
  const int NCHUNK = 9 * NCB;
  __shared__ h16 sh[3 * 66 * CIN];

  int bid = blockIdx.x;
  int wt = bid & 1;          // W/64 = 2 segments
  int t  = bid >> 1;
  int y0 = t & 127;
  int bb = t >> 7;
  int w0 = wt * 64;
  int tid = threadIdx.x;

  int lane = tid & 31;
  int cot  = tid >> 5;                 // wave id = C_out tile
  const v16h* afrag = (const v16h*)wp + ((size_t)cot * NCHUNK) * 32 + lane;

  // warm the weight stream for this wave while we stage the input patch
  {
    const char* wbase = (const char*)((const v16h*)wp + (size_t)cot * NCHUNK * 32);
    for (int q = lane; q < NCHUNK; q += 32)
      __builtin_prefetch(wbase + (size_t)q * 1024, 0, 0);   // global_prefetch_b8
  }

  // ---- stage 3 rows x 66 px x CIN (zero-padded) into LDS
  for (int i = tid; i < 3 * 66 * CIN; i += blockDim.x) {
    int r = i / (66 * CIN);
    int rem = i - r * 66 * CIN;
    int px = rem / CIN;
    int ci = rem - px * CIN;
    int y = y0 + r - 1;
    int xw = w0 + px - 1;
    h16 v = (h16)0.f;
    if (y >= 0 && y < HH && xw >= 0 && xw < WW)
      v = xh[(((size_t)bb * CIN + ci) * HH + y) * WW + xw];
    sh[i] = v;
  }
  __syncthreads();

  int n = lane & 15, hh = lane >> 4;

  v8f acc0 = {}, acc1 = {}, acc2 = {}, acc3 = {};

#pragma unroll 1
  for (int koff = 0; koff < 9; koff++) {
    int kh = koff / 3, kw = koff - kh * 3;
    const h16* brow = sh + (kh * 66 + n + kw) * CIN + hh * 8;
    for (int cib = 0; cib < NCB; cib++) {
      v16h a = afrag[(koff * NCB + cib) * 32];
      const h16* bp = brow + cib * 32;
      v16h b0, b1, b2, b3;
#pragma unroll
      for (int i = 0; i < 16; i++) {
        b0[i] = bp[(i < 8 ? i : i + 8)];
        b1[i] = bp[16 * CIN + (i < 8 ? i : i + 8)];
        b2[i] = bp[32 * CIN + (i < 8 ? i : i + 8)];
        b3[i] = bp[48 * CIN + (i < 8 ? i : i + 8)];
      }
      acc0 = __builtin_amdgcn_wmma_f32_16x16x32_f16(false, a, false, b0, (short)0, acc0, false, false);
      acc1 = __builtin_amdgcn_wmma_f32_16x16x32_f16(false, a, false, b1, (short)0, acc1, false, false);
      acc2 = __builtin_amdgcn_wmma_f32_16x16x32_f16(false, a, false, b2, (short)0, acc2, false, false);
      acc3 = __builtin_amdgcn_wmma_f32_16x16x32_f16(false, a, false, b3, (short)0, acc3, false, false);
    }
  }

  // ---- D layout: VGPR r, lanes0-15 -> M=r, lanes16-31 -> M=8+r
#pragma unroll
  for (int r = 0; r < 8; r++) {
    int co = cot * 16 + r + 8 * hh;
    float bv = bias[co];
    size_t base = (((size_t)bb * COUT + co) * HH + y0) * WW + w0 + n;
    outp[base +  0] = acc0[r] + bv;
    outp[base + 16] = acc1[r] + bv;
    outp[base + 32] = acc2[r] + bv;
    outp[base + 48] = acc3[r] + bv;
  }
}

// -------------------------------------------------- BN stats (per channel)
template<int C>
__global__ void k_bn_stats(const float* __restrict__ t,
                           float* __restrict__ mean, float* __restrict__ rvar) {
  int ch = blockIdx.x;
  int tid = threadIdx.x;
  const size_t NPC = (size_t)BSZ * HWT;
  float s = 0.f, s2 = 0.f;
  for (size_t i = tid; i < NPC; i += 256) {
    size_t b = i >> 14;
    size_t p = i & (HWT - 1);
    float v = t[(b * C + ch) * HWT + p];
    s += v; s2 += v * v;
  }
  __shared__ float rs[256], rs2[256];
  rs[tid] = s; rs2[tid] = s2;
  __syncthreads();
  for (int st = 128; st > 0; st >>= 1) {
    if (tid < st) { rs[tid] += rs[tid + st]; rs2[tid] += rs2[tid + st]; }
    __syncthreads();
  }
  if (tid == 0) {
    float m = rs[0] / (float)NPC;
    float v = rs2[0] / (float)NPC - m * m;
    mean[ch] = m;
    rvar[ch] = rsqrtf(v + EPSV);
  }
}

// ------------------------------------------- BN + ReLU -> f16 activation
template<int C>
__global__ void k_bn_relu_h(const float* __restrict__ t, const float* __restrict__ mean,
                            const float* __restrict__ rvar, const float* __restrict__ g,
                            const float* __restrict__ be, h16* __restrict__ o, size_t nn) {
  size_t i = (size_t)blockIdx.x * blockDim.x + threadIdx.x;
  if (i >= nn) return;
  int c = (int)((i >> 14) % C);
  float v = (t[i] - mean[c]) * rvar[c] * g[c] + be[c];
  o[i] = (h16)fmaxf(v, 0.f);
}

// ------------------- BN (no relu) + transpose to [B][HW][32] for kmeans
__global__ void k_bn_tr(const float* __restrict__ t, const float* __restrict__ mean,
                        const float* __restrict__ rvar, const float* __restrict__ g,
                        const float* __restrict__ be, float* __restrict__ fT) {
  size_t i = (size_t)blockIdx.x * blockDim.x + threadIdx.x;
  const size_t nn = (size_t)BSZ * 32 * HWT;
  if (i >= nn) return;
  int c = (int)((i >> 14) & 31);
  size_t b = i >> 19;            // / (32*HW)
  size_t p = i & (HWT - 1);
  float v = (t[i] - mean[c]) * rvar[c] * g[c] + be[c];
  fT[(b * HWT + p) * 32 + c] = v;
}

// --------------------------------------------------------------- K-means
// cent: [B][KC][32] f32 in global; sums: [B][KC][33] f32 in global.
__global__ void k_km_init(const float* __restrict__ fT, float* __restrict__ cent,
                          float* __restrict__ sums) {
  int b = blockIdx.x;
  int tid = threadIdx.x;
  if (tid < KC * 32) {
    int k = tid >> 5, c = tid & 31;
    int idx = (k * (HWT - 1)) / 4;               // linspace(0,16383,5) -> int32
    cent[b * (KC * 32) + tid] = fT[((size_t)b * HWT + idx) * 32 + c];
  }
  if (tid < KC * 33) sums[b * (KC * 33) + tid] = 0.f;
}

// grid = B * 16 chunks; each block handles 1024 points with LDS-privatized bins
__global__ void __launch_bounds__(256)
k_km_assign(const float* __restrict__ fT, const float* __restrict__ cent,
            float* __restrict__ sums, unsigned char* __restrict__ lbl, int write_lbl) {
  int blk = blockIdx.x;
  int b = blk >> 4;
  int chk = blk & 15;
  int tid = threadIdx.x;
  const float* P = fT + (size_t)b * HWT * 32;
  __shared__ float sc[KC][32];
  __shared__ float part[KC][33];
  if (tid < KC * 32) ((float*)sc)[tid] = cent[b * (KC * 32) + tid];
  for (int i = tid; i < KC * 33; i += 256) ((float*)part)[i] = 0.f;
  __syncthreads();

  float cn[KC];
#pragma unroll
  for (int k = 0; k < KC; k++) {
    float s = 0.f;
    for (int c = 0; c < 32; c++) { float cc = sc[k][c]; s += cc * cc; }
    cn[k] = s;
  }

  int p0 = chk * 1024;
  for (int pp = p0 + tid; pp < p0 + 1024; pp += 256) {
    const float* pt = P + (size_t)pp * 32;
    float pv[32];
#pragma unroll
    for (int c = 0; c < 32; c++) pv[c] = pt[c];
    float d[KC];
#pragma unroll
    for (int k = 0; k < KC; k++) d[k] = cn[k];
    for (int c = 0; c < 32; c++) {
      float x = pv[c];
#pragma unroll
      for (int k = 0; k < KC; k++) d[k] -= 2.f * x * sc[k][c];
    }
    int l = 0; float best = d[0];
#pragma unroll
    for (int k = 1; k < KC; k++) if (d[k] < best) { best = d[k]; l = k; }
    if (write_lbl) lbl[(size_t)b * HWT + pp] = (unsigned char)l;
    for (int c = 0; c < 32; c++) atomicAdd(&part[l][c], pv[c]);
    atomicAdd(&part[l][32], 1.f);
  }
  __syncthreads();
  if (tid < KC * 33) atomicAdd(&sums[b * (KC * 33) + tid], ((float*)part)[tid]);
}

// centroid update (or final means); also re-zeros sums for the next round
__global__ void k_km_update(float* __restrict__ cent, float* __restrict__ sums, int final_pass) {
  int b = blockIdx.x;
  int tid = threadIdx.x;
  __shared__ float ss[KC * 33];
  if (tid < KC * 33) ss[tid] = sums[b * (KC * 33) + tid];
  __syncthreads();
  if (tid < KC * 32) {
    int k = tid >> 5, c = tid & 31;
    float cnt = ss[k * 33 + 32];
    float s = ss[k * 33 + c];
    if (final_pass)          cent[b * (KC * 32) + tid] = s / fmaxf(cnt, 1.f);
    else if (cnt > 0.f)      cent[b * (KC * 32) + tid] = s / cnt;
  }
  if (tid < KC * 33) sums[b * (KC * 33) + tid] = 0.f;
}

// scatter-mean writeback: out[b][c][p] = means[lbl[p]][c]
__global__ void __launch_bounds__(256)
k_km_out(const float* __restrict__ cent, const unsigned char* __restrict__ lbl,
         float* __restrict__ out) {
  int blk = blockIdx.x;
  int b = blk >> 4;
  int chk = blk & 15;
  int tid = threadIdx.x;
  __shared__ float cl[KC * 32];
  if (tid < KC * 32) cl[tid] = cent[b * (KC * 32) + tid];
  __syncthreads();
  int p0 = chk * 1024;
  for (int pp = p0 + tid; pp < p0 + 1024; pp += 256) {
    int l = lbl[(size_t)b * HWT + pp];
#pragma unroll
    for (int c = 0; c < 32; c++)
      out[((size_t)b * 32 + c) * HWT + pp] = cl[l * 32 + c];
  }
}

// ---------------------------------------------------------------- launcher
extern "C" void kernel_launch(void* const* d_in, const int* in_sizes, int n_in,
                              void* d_out, int out_size, void* d_ws, size_t ws_size,
                              hipStream_t stream) {
  (void)in_sizes; (void)n_in; (void)out_size; (void)ws_size;
  const float* x   = (const float*)d_in[0];
  const float* w1  = (const float*)d_in[1];
  const float* b1  = (const float*)d_in[2];
  const float* g1  = (const float*)d_in[3];
  const float* be1 = (const float*)d_in[4];
  const float* w2  = (const float*)d_in[5];
  const float* b2  = (const float*)d_in[6];
  const float* g2  = (const float*)d_in[7];
  const float* be2 = (const float*)d_in[8];
  const float* w3  = (const float*)d_in[9];
  const float* b3  = (const float*)d_in[10];
  const float* g3  = (const float*)d_in[11];
  const float* be3 = (const float*)d_in[12];
  float* out = (float*)d_out;

  char* ws = (char*)d_ws;
  size_t off = 0;
  auto alloc = [&](size_t bytes) -> void* {
    void* p = ws + off;
    off += (bytes + 255) & ~(size_t)255;
    return p;
  };

  const size_t n128 = (size_t)BSZ * 128 * HWT;
  const size_t n64  = (size_t)BSZ * 64  * HWT;
  const size_t n32  = (size_t)BSZ * 32  * HWT;

  h16*   xh    = (h16*)  alloc(n128 * 2);
  h16*   h1h   = (h16*)  alloc(n128 * 2);
  h16*   h2h   = (h16*)  alloc(n64  * 2);
  float* tmp   = (float*)alloc(n128 * 4);
  float* fT    = (float*)alloc(n32  * 4);
  h16*   wp1   = (h16*)  alloc((size_t)8 * 36 * 32 * 16 * 2);
  h16*   wp2   = (h16*)  alloc((size_t)4 * 36 * 32 * 16 * 2);
  h16*   wp3   = (h16*)  alloc((size_t)2 * 18 * 32 * 16 * 2);
  float* mean  = (float*)alloc(128 * 4);
  float* rvar  = (float*)alloc(128 * 4);
  float* centg = (float*)alloc((size_t)BSZ * KC * 32 * 4);
  float* sumsg = (float*)alloc((size_t)BSZ * KC * 33 * 4);
  unsigned char* lblb = (unsigned char*)alloc((size_t)BSZ * HWT);

  const int CONV_BLOCKS = BSZ * HH * (WW / 64);  // 2048

  // ---- layer 1: 128 -> 128
  k_cvt_f16<<<(int)((n128 + 255) / 256), 256, 0, stream>>>(x, xh, (int)n128);
  k_pack_w<128,128><<<(8 * 36 * 32 + 255) / 256, 256, 0, stream>>>(w1, wp1);
  k_conv<128,128><<<CONV_BLOCKS, 256, 0, stream>>>(xh, wp1, b1, tmp);
  k_bn_stats<128><<<128, 256, 0, stream>>>(tmp, mean, rvar);
  k_bn_relu_h<128><<<(int)((n128 + 255) / 256), 256, 0, stream>>>(tmp, mean, rvar, g1, be1, h1h, n128);

  // ---- layer 2: 128 -> 64
  k_pack_w<128,64><<<(4 * 36 * 32 + 255) / 256, 256, 0, stream>>>(w2, wp2);
  k_conv<128,64><<<CONV_BLOCKS, 128, 0, stream>>>(h1h, wp2, b2, tmp);
  k_bn_stats<64><<<64, 256, 0, stream>>>(tmp, mean, rvar);
  k_bn_relu_h<64><<<(int)((n64 + 255) / 256), 256, 0, stream>>>(tmp, mean, rvar, g2, be2, h2h, n64);

  // ---- layer 3: 64 -> 32 (no relu, transpose for kmeans)
  k_pack_w<64,32><<<(2 * 18 * 32 + 255) / 256, 256, 0, stream>>>(w3, wp3);
  k_conv<64,32><<<CONV_BLOCKS, 64, 0, stream>>>(h2h, wp3, b3, tmp);
  k_bn_stats<32><<<32, 256, 0, stream>>>(tmp, mean, rvar);
  k_bn_tr<<<(int)((n32 + 255) / 256), 256, 0, stream>>>(tmp, mean, rvar, g3, be3, fT);

  // ---- per-batch kmeans (Lloyd x10) + final means + scatter-mean
  k_km_init<<<BSZ, 256, 0, stream>>>(fT, centg, sumsg);
  for (int it = 0; it < KITERS; it++) {
    k_km_assign<<<BSZ * 16, 256, 0, stream>>>(fT, centg, sumsg, lblb, 0);
    k_km_update<<<BSZ, 256, 0, stream>>>(centg, sumsg, 0);
  }
  k_km_assign<<<BSZ * 16, 256, 0, stream>>>(fT, centg, sumsg, lblb, 1);
  k_km_update<<<BSZ, 256, 0, stream>>>(centg, sumsg, 1);
  k_km_out<<<BSZ * 16, 256, 0, stream>>>(centg, lblb, out);
}
